// Quantize_5669356832545
// MI455X (gfx1250) — compile-verified, matrix operand order checked by reading, not run on hardware
//
#include <hip/hip_runtime.h>
#include <hip/hip_bf16.h>

// ---- CDNA5 WMMA fragment types (wave32) ----
typedef __attribute__((ext_vector_type(16))) __bf16 v16bf;
typedef __attribute__((ext_vector_type(8)))  __bf16 v8bf;
typedef __attribute__((ext_vector_type(8)))  float  v8f;
typedef __attribute__((ext_vector_type(4)))  unsigned int u32x4;
typedef __attribute__((ext_vector_type(8)))  int i32x8;
typedef __attribute__((ext_vector_type(4)))  int i32x4;

#define N_ROWS   131072     // 32*64*64
#define DIM      64
#define NE       1024
#define DECAY_F  0.99f
#define EPS_F    1e-5f

// ---------------- workspace layout (bytes) ----------------
// DMA region (contiguous, one TDM descriptor):
//   EhiT : bf16 [NE][DIM] @ 0        (131072 B)  hi(-2*E), code-major
//   EloT : bf16 [NE][DIM] @ 131072   (131072 B)  lo(-2*E)
//   ee   : f32  [NE]      @ 262144   (4096 B)    ||e_k||^2
// ET     : f32  [NE][DIM] @ 266240   (262144 B)  E transposed (gather)
// counts : f32  [NE]      @ 528384   (4096 B)
// esumT  : f32  [NE][DIM] @ 532480   (262144 B)
// sse    : f32  [1]       @ 794624
#define WS_EHIT   0
#define WS_ELOT   131072
#define WS_EE     262144
#define WS_ET     266240
#define WS_CNT    528384
#define WS_ESUM   532480
#define WS_SSE    794624

#define LDS_BYTES 266240            // mirrors the DMA region layout
#define TDM_ELEMS (266240 / 8)      // 33280 x 8-byte elements

__global__ __launch_bounds__(256) void vq_zero(float* __restrict__ p, int n) {
    int i = blockIdx.x * 256 + threadIdx.x;
    if (i < n) p[i] = 0.0f;
}

// Transpose codebook, split -2*E into bf16 hi/lo, compute ||e_k||^2.
__global__ __launch_bounds__(256) void vq_prep(const float* __restrict__ embed,
                                               unsigned short* __restrict__ EhiT_u,
                                               unsigned short* __restrict__ EloT_u,
                                               float* __restrict__ ET,
                                               float* __restrict__ ee) {
    int k = blockIdx.x * 256 + threadIdx.x;
    if (k >= NE) return;
    __bf16* EhiT = (__bf16*)EhiT_u;
    __bf16* EloT = (__bf16*)EloT_u;
    float s = 0.0f;
    #pragma unroll
    for (int d = 0; d < DIM; ++d) {
        float v = embed[(size_t)d * NE + k];   // coalesced across k
        s += v * v;
        ET[(size_t)k * DIM + d] = v;
        float m2 = -2.0f * v;
        __bf16 h = (__bf16)m2;
        EhiT[(size_t)k * DIM + d] = h;
        EloT[(size_t)k * DIM + d] = (__bf16)(m2 - (float)h);
    }
    ee[k] = s;
}

// 512 threads = 16 waves per block; one wave per 16-row tile.
// Codebook (bf16 hi/lo of -2*E) + ||e||^2 DMA'd to LDS by the Tensor Data Mover
// (wave 0 issues the descriptor; other waves prep A fragments concurrently);
// inner loop: ds_load_b128 B fragments + 6x v_wmma_f32_16x16x32_bf16 per 16-code tile.
__global__ __launch_bounds__(512) void vq_main(
    const float* __restrict__ X,
    const unsigned short* __restrict__ EhiT_u,   // hi|lo|ee contiguous: 266240 B
    const float* __restrict__ ET,
    float* __restrict__ out_q,
    float* __restrict__ out_ind,
    float* __restrict__ counts,
    float* __restrict__ esumT,
    float* __restrict__ sse)
{
    extern __shared__ float4 smem4[];            // 16B-aligned dynamic LDS
    char* smem = (char*)smem4;

    const int lane = threadIdx.x & 31;
    const int wave = threadIdx.x >> 5;

#if defined(__has_builtin) && __has_builtin(__builtin_amdgcn_tensor_load_to_lds)
    // ---- TDM: DMA 266240 B codebook region -> LDS (one descriptor, wave 0) ----
    if (wave == 0) {
        unsigned long long gaddr = (unsigned long long)(uintptr_t)EhiT_u;
        unsigned int lds_base = (unsigned int)(uintptr_t)smem;  // low 32b = LDS byte addr

        u32x4 g0;                               // D# group 0 (ISA 8.3)
        g0[0] = 1u;                             // count=1, user descriptor
        g0[1] = lds_base;                       // lds_addr
        g0[2] = (unsigned int)gaddr;            // global_addr[31:0]
        g0[3] = ((unsigned int)(gaddr >> 32) & 0x01FFFFFFu) | 0x80000000u; // addr[56:32], type=2

        union { unsigned int d[8]; unsigned long long q[4]; } g1u;
        g1u.q[0] = g1u.q[1] = g1u.q[2] = g1u.q[3] = 0ull;
        g1u.q[0] |= 3ull << 16;                                  // data_size = 8 B
        g1u.q[0] |= (unsigned long long)(TDM_ELEMS & 0xFFFF) << 48; // tensor_dim0[15:0]
        g1u.q[1] |= (unsigned long long)(TDM_ELEMS >> 16);          // tensor_dim0[31:16]
        g1u.q[1] |= 1ull << 16;                                  // tensor_dim1 = 1
        g1u.q[1] |= (unsigned long long)(TDM_ELEMS & 0xFFFF) << 48; // tile_dim0
        g1u.q[2] |= 1ull;                                        // tile_dim1 = 1
        g1u.q[2] |= (unsigned long long)TDM_ELEMS << 32;         // tensor_dim0_stride[31:0]
        i32x8 g1;
        #pragma unroll
        for (int i = 0; i < 8; ++i) g1[i] = (int)g1u.d[i];

        i32x4 z4 = {0, 0, 0, 0};                // 1-D/2-D tensor: groups 2,3 unused
#if __clang_major__ >= 23
        i32x8 z8 = {0, 0, 0, 0, 0, 0, 0, 0};
        __builtin_amdgcn_tensor_load_to_lds(g0, g1, z4, z4, z8, 0);
#else
        __builtin_amdgcn_tensor_load_to_lds(g0, g1, z4, z4, 0);
#endif
        __builtin_amdgcn_s_wait_tensorcnt(0);   // TENSORcnt -> 0: DMA complete
    }
#else
    // ---- fallback: cooperative register-roundtrip copy ----
    {
        const float4* src = (const float4*)EhiT_u;   // 16640 float4
        for (int i = threadIdx.x; i < LDS_BYTES / 16; i += 512)
            smem4[i] = src[i];
    }
#endif

    const int tile = blockIdx.x * 16 + wave;   // 512 blocks * 16 waves = 8192 tiles
    const int m0   = tile << 4;                // first of 16 rows
    const int half = lane >> 4;
    const int sub  = lane & 15;

    // ---- A fragments: 16 rows x K=64, per ISA 16-bit A 16x32 layout ----
    // (overlaps with the TDM transfer; barrier below publishes LDS)
    v16bf Ahi[2], Alo[2];
    {
        const float* xrow = X + (size_t)(m0 + sub) * DIM;
        #pragma unroll
        for (int f = 0; f < 2; ++f) {
            const int kb = f * 32 + half * 8;
            float tmp[16] __attribute__((aligned(16)));
            *(float4*)&tmp[0]  = *(const float4*)(xrow + kb);
            *(float4*)&tmp[4]  = *(const float4*)(xrow + kb + 4);
            *(float4*)&tmp[8]  = *(const float4*)(xrow + kb + 16);
            *(float4*)&tmp[12] = *(const float4*)(xrow + kb + 20);
            #pragma unroll
            for (int i = 0; i < 16; ++i) {
                __bf16 h = (__bf16)tmp[i];
                Ahi[f][i] = h;
                Alo[f][i] = (__bf16)(tmp[i] - (float)h);
            }
        }
    }

    __syncthreads();

    const __bf16* ldsHi = (const __bf16*)smem;            // [NE][DIM]
    const __bf16* ldsLo = ldsHi + (size_t)NE * DIM;       // [NE][DIM]
    const float*  ldsEE = (const float*)(smem + 262144);  // [NE]

    float minv[8];
    int   mink[8];
    #pragma unroll
    for (int j = 0; j < 8; ++j) { minv[j] = 3.4e38f; mink[j] = 0; }

    for (int n0 = 0; n0 < NE; n0 += 16) {
        // B fragment from LDS: lane holds column N=n0+sub; K chunking mirrors A.
        const __bf16* eh = ldsHi + (size_t)(n0 + sub) * DIM + half * 8;
        const __bf16* el = ldsLo + (size_t)(n0 + sub) * DIM + half * 8;
        v8bf h0 = *(const v8bf*)(eh);
        v8bf h1 = *(const v8bf*)(eh + 16);
        v8bf h2 = *(const v8bf*)(eh + 32);
        v8bf h3 = *(const v8bf*)(eh + 48);
        v8bf l0 = *(const v8bf*)(el);
        v8bf l1 = *(const v8bf*)(el + 16);
        v8bf l2 = *(const v8bf*)(el + 32);
        v8bf l3 = *(const v8bf*)(el + 48);
        v16bf Bhi0 = __builtin_shufflevector(h0, h1, 0,1,2,3,4,5,6,7,8,9,10,11,12,13,14,15);
        v16bf Bhi1 = __builtin_shufflevector(h2, h3, 0,1,2,3,4,5,6,7,8,9,10,11,12,13,14,15);
        v16bf Blo0 = __builtin_shufflevector(l0, l1, 0,1,2,3,4,5,6,7,8,9,10,11,12,13,14,15);
        v16bf Blo1 = __builtin_shufflevector(l2, l3, 0,1,2,3,4,5,6,7,8,9,10,11,12,13,14,15);

        // C init = ||e_col||^2 ; then accumulate -2 x.e via compensated bf16 products
        v8f c;
        const float eecol = ldsEE[n0 + sub];
        #pragma unroll
        for (int j = 0; j < 8; ++j) c[j] = eecol;

        c = __builtin_amdgcn_wmma_f32_16x16x32_bf16(false, Ahi[0], false, Bhi0, (short)0, c, false, false);
        c = __builtin_amdgcn_wmma_f32_16x16x32_bf16(false, Ahi[0], false, Blo0, (short)0, c, false, false);
        c = __builtin_amdgcn_wmma_f32_16x16x32_bf16(false, Alo[0], false, Bhi0, (short)0, c, false, false);
        c = __builtin_amdgcn_wmma_f32_16x16x32_bf16(false, Ahi[1], false, Bhi1, (short)0, c, false, false);
        c = __builtin_amdgcn_wmma_f32_16x16x32_bf16(false, Ahi[1], false, Blo1, (short)0, c, false, false);
        c = __builtin_amdgcn_wmma_f32_16x16x32_bf16(false, Alo[1], false, Bhi1, (short)0, c, false, false);

        const int col = n0 + sub;
        #pragma unroll
        for (int j = 0; j < 8; ++j) {        // branchless -> v_cndmask, EXEC untouched
            bool m = c[j] < minv[j];
            minv[j] = m ? c[j] : minv[j];
            mink[j] = m ? col  : mink[j];
        }
    }

    // ---- argmin across the 16 lanes of each half (ties -> lowest index) ----
    #pragma unroll
    for (int off = 1; off < 16; off <<= 1) {
        #pragma unroll
        for (int j = 0; j < 8; ++j) {
            float ov = __shfl_xor(minv[j], off, 32);
            int   ok = __shfl_xor(mink[j], off, 32);
            bool take = (ov < minv[j]) || (ov == minv[j] && ok < mink[j]);
            minv[j] = take ? ov : minv[j];
            mink[j] = take ? ok : mink[j];
        }
    }

    // ---- gather codewords, ST output, EMA scatter stats, SSE ----
    float sacc = 0.0f;
    #pragma unroll
    for (int j = 0; j < 8; ++j) {
        const int r  = m0 + half * 8 + j;     // half 0 -> rows 0-7, half 1 -> rows 8-15
        const int k  = mink[j];               // uniform within each half
        const int d0 = sub * 4;               // 16 lanes x 4 floats = full 64-dim row
        float4 q  = *(const float4*)(ET + (size_t)k * DIM + d0);
        float4 xv = *(const float4*)(X  + (size_t)r * DIM + d0);
        float4 o;                             // straight-through: x + (q - x), f32 exact path
        o.x = xv.x + (q.x - xv.x);
        o.y = xv.y + (q.y - xv.y);
        o.z = xv.z + (q.z - xv.z);
        o.w = xv.w + (q.w - xv.w);
        *(float4*)(out_q + (size_t)r * DIM + d0) = o;
        float dx = q.x - xv.x, dy = q.y - xv.y, dz = q.z - xv.z, dw = q.w - xv.w;
        sacc += dx * dx + dy * dy + dz * dz + dw * dw;
        atomicAdd(esumT + (size_t)k * DIM + d0 + 0, xv.x);   // coalesced f32 atomics
        atomicAdd(esumT + (size_t)k * DIM + d0 + 1, xv.y);
        atomicAdd(esumT + (size_t)k * DIM + d0 + 2, xv.z);
        atomicAdd(esumT + (size_t)k * DIM + d0 + 3, xv.w);
        if (sub == 0) {
            out_ind[r] = (float)k;
            atomicAdd(counts + k, 1.0f);
        }
    }
    #pragma unroll
    for (int off = 1; off < 32; off <<= 1) sacc += __shfl_xor(sacc, off, 32);
    if (lane == 0) atomicAdd(sse, sacc);
}

// EMA update + normalization + diff. One 1024-thread block (k = code index).
__global__ __launch_bounds__(1024) void vq_finalize(
    const float* __restrict__ cluster_size,
    const float* __restrict__ embed_avg,
    const float* __restrict__ counts,
    const float* __restrict__ esumT,
    const float* __restrict__ sse,
    float* __restrict__ out_diff,
    float* __restrict__ out_ncs,
    float* __restrict__ out_avg,
    float* __restrict__ out_embed)
{
    __shared__ float red[1024];
    const int k = threadIdx.x;
    float ncs = cluster_size[k] * DECAY_F + (1.0f - DECAY_F) * counts[k];
    out_ncs[k] = ncs;
    red[k] = ncs;
    __syncthreads();
    #pragma unroll
    for (int s = 512; s > 0; s >>= 1) {
        if (k < s) red[k] += red[k + s];
        __syncthreads();
    }
    const float n  = red[0];
    const float cs = (ncs + EPS_F) / (n + (float)NE * EPS_F) * n;
    const float inv = 1.0f / cs;
    #pragma unroll
    for (int d = 0; d < DIM; ++d) {
        float ea = embed_avg[(size_t)d * NE + k] * DECAY_F +
                   (1.0f - DECAY_F) * esumT[(size_t)k * DIM + d];
        out_avg[(size_t)d * NE + k]   = ea;      // coalesced across k
        out_embed[(size_t)d * NE + k] = ea * inv;
    }
    if (k == 0) out_diff[0] = sse[0] * (1.0f / (float)(N_ROWS * DIM));
}

extern "C" void kernel_launch(void* const* d_in, const int* in_sizes, int n_in,
                              void* d_out, int out_size, void* d_ws, size_t ws_size,
                              hipStream_t stream) {
    const float* X       = (const float*)d_in[0];   // [32,64,64,64]
    const float* embed   = (const float*)d_in[1];   // [64,1024]
    const float* cluster = (const float*)d_in[2];   // [1024]
    const float* eavg    = (const float*)d_in[3];   // [64,1024]

    char* ws = (char*)d_ws;
    unsigned short* EhiT = (unsigned short*)(ws + WS_EHIT);
    unsigned short* EloT = (unsigned short*)(ws + WS_ELOT);
    float* ee     = (float*)(ws + WS_EE);
    float* ET     = (float*)(ws + WS_ET);
    float* counts = (float*)(ws + WS_CNT);
    float* esumT  = (float*)(ws + WS_ESUM);
    float* sseacc = (float*)(ws + WS_SSE);

    float* out       = (float*)d_out;
    float* out_q     = out;                         // 8388608
    float* out_diff  = out + 8388608;               // 1
    float* out_ind   = out + 8388609;               // 131072
    float* out_ncs   = out + 8388609 + 131072;      // 1024
    float* out_avg   = out_ncs + 1024;              // 65536
    float* out_embed = out_avg + 65536;             // 65536

    // counts + esumT + sse are contiguous: zero them in one pass (1024+65536+1)
    vq_zero<<<(66561 + 255) / 256, 256, 0, stream>>>(counts, 66561);
    vq_prep<<<4, 256, 0, stream>>>(embed, EhiT, EloT, ET, ee);
    vq_main<<<512, 512, LDS_BYTES, stream>>>(X, EhiT, ET,
                                             out_q, out_ind, counts, esumT, sseacc);
    vq_finalize<<<1, 1024, 0, stream>>>(cluster, eavg, counts, esumT, sseacc,
                                        out_diff, out_ncs, out_avg, out_embed);
}
